// AttentionBlock_6433861009457
// MI455X (gfx1250) — matrix operand order repaired and checked by reading, not compiled, not run on hardware
//
#include <hip/hip_runtime.h>
#include <math.h>

#define BATCH 8
#define C 256
#define HW 2304
#define NH 4
#define HD 64
#define GROUPS 32
#define CPG 8
#define EPS 1e-5f

typedef __attribute__((ext_vector_type(16))) _Float16 v16h;
typedef __attribute__((ext_vector_type(8)))  _Float16 v8h;
typedef __attribute__((ext_vector_type(8)))  float    v8f;

static __device__ __forceinline__ v16h cat8(v8h lo, v8h hi) {
    return __builtin_shufflevector(lo, hi, 0,1,2,3,4,5,6,7,8,9,10,11,12,13,14,15);
}

static __device__ __forceinline__ v8f wmma_f16(v16h a, v16h b, v8f c) {
    // D = A(16x32 f16) * B(32x16 f16) + C(16x16 f32)
    return __builtin_amdgcn_wmma_f32_16x16x32_f16(false, a, false, b, (short)0, c, false, false);
}

// Async 16B/lane copy global -> LDS, tracked by ASYNCcnt (CDNA5 ISA).
static __device__ __forceinline__ void async_cp16(unsigned lds_addr, const void* gaddr) {
    asm volatile("global_load_async_to_lds_b128 %0, %1, off"
                 :: "v"(lds_addr), "v"(gaddr) : "memory");
}

// ---------------------------------------------------------------------------
// 1) GroupNorm statistics: one block per (b, group); group data is contiguous
// ---------------------------------------------------------------------------
__global__ void __launch_bounds__(256) k_gn_stats(const float* __restrict__ x,
                                                  float* __restrict__ mean,
                                                  float* __restrict__ rstd) {
    __shared__ float ss[256], sq[256];
    const int bg = blockIdx.x;                       // b*32 + g, 0..255
    const float* p = x + (size_t)bg * (CPG * HW);    // contiguous 8*2304 floats
    float s = 0.f, q = 0.f;
    for (int i = threadIdx.x; i < CPG * HW; i += 256) {
        float v = p[i];
        s += v; q += v * v;
    }
    ss[threadIdx.x] = s; sq[threadIdx.x] = q;
    __syncthreads();
    for (int off = 128; off > 0; off >>= 1) {
        if ((int)threadIdx.x < off) {
            ss[threadIdx.x] += ss[threadIdx.x + off];
            sq[threadIdx.x] += sq[threadIdx.x + off];
        }
        __syncthreads();
    }
    if (threadIdx.x == 0) {
        const float inv_n = 1.0f / (float)(CPG * HW);
        float m = ss[0] * inv_n;
        float var = sq[0] * inv_n - m * m;
        mean[bg] = m;
        rstd[bg] = rsqrtf(var + EPS);
    }
}

// ---------------------------------------------------------------------------
// 2) Normalize + affine, write f16 in pixel-major layout xn[b][p][c]
// ---------------------------------------------------------------------------
__global__ void __launch_bounds__(256) k_gn_norm(const float* __restrict__ x,
                                                 const float* __restrict__ mean,
                                                 const float* __restrict__ rstd,
                                                 const float* __restrict__ gw,
                                                 const float* __restrict__ gb,
                                                 _Float16* __restrict__ xn) {
    const int idx = blockIdx.x * 256 + threadIdx.x;  // < BATCH*C*HW
    const int b = idx / (C * HW);
    const int rem = idx - b * C * HW;
    const int c = rem / HW;
    const int p = rem - c * HW;
    const int sg = b * GROUPS + (c >> 3);
    float v = (x[idx] - mean[sg]) * rstd[sg] * gw[c] + gb[c];
    xn[((size_t)(b * HW + p) << 8) + c] = (_Float16)v;
}

// ---------------------------------------------------------------------------
// 3) fp32 -> fp16 weight conversion
// ---------------------------------------------------------------------------
__global__ void __launch_bounds__(256) k_w2h(const float* __restrict__ w,
                                             _Float16* __restrict__ wh, int n) {
    int i = blockIdx.x * 256 + threadIdx.x;
    if (i < n) wh[i] = (_Float16)w[i];
}

// ---------------------------------------------------------------------------
// 4) QKV GEMM: C(768x2304) = W(768x256) x Xn(256x2304) per batch.
//    One 16x16 tile per wave. Epilogue scatters into attention-friendly
//    layouts: q,k -> [bh][p][d] (rows contiguous), v -> [bh][d][p] (transposed)
// ---------------------------------------------------------------------------
__global__ void __launch_bounds__(256) k_qkv(const _Float16* __restrict__ wh,
                                             const float* __restrict__ bias,
                                             const _Float16* __restrict__ xn,
                                             _Float16* __restrict__ q,
                                             _Float16* __restrict__ k,
                                             _Float16* __restrict__ v) {
    const int t = blockIdx.x * 8 + (threadIdx.x >> 5);   // wave tile id < 55296
    const int b = t / (48 * 144);
    const int r = t - b * 48 * 144;
    const int o0 = (r / 144) * 16;
    const int p0 = (r % 144) * 16;
    const int l = threadIdx.x & 31;
    const int n16 = l & 15;
    const bool hi = l >= 16;
    const int kbA = hi ? 16 : 0;
    const int kbB = hi ? 16 : 0;

    const _Float16* xb = xn + ((size_t)(b * HW + p0 + n16) << 8);
    v8f acc = {};
    for (int k0 = 0; k0 < C; k0 += 32) {
        const _Float16* ap = wh + (size_t)(o0 + n16) * C + k0 + kbA;
        v16h a = cat8(*(const v8h*)ap, *(const v8h*)(ap + 8));
        v16h bb = *(const v16h*)(xb + k0 + kbB);
        acc = wmma_f16(a, bb, acc);
    }
    const int rowoff = hi ? 8 : 0;
#pragma unroll
    for (int rr = 0; rr < 8; ++rr) {
        const int o = o0 + rr + rowoff;
        const float val = acc[rr] + bias[o];
        const int p = p0 + n16;
        const int which = o >> 8;       // 0=q 1=k 2=v
        const int oc = o & 255;
        const int head = oc >> 6, d = oc & 63;
        const size_t bh = (size_t)(b * NH + head);
        if (which == 0)      q[(bh * HW + p) * HD + d] = (_Float16)(val * 0.125f); // fold 1/sqrt(64)
        else if (which == 1) k[(bh * HW + p) * HD + d] = (_Float16)val;
        else                 v[(bh * HD + d) * HW + p] = (_Float16)val;
    }
}

// ---------------------------------------------------------------------------
// 5) Flash attention. 4 waves/block, all on the same (b,head) (144 % 4 == 0),
//    each wave owns a 16-query tile. Per 32-key block:
//      - K tile (32x64 f16, 4KB) and V^T tile (64x32 f16, 4KB) staged into
//        LDS cooperatively via global_load_async_to_lds_b128, double-buffered
//        so the copy of block j+1 overlaps compute of block j (ASYNCcnt,
//        in-order completion => s_wait_asynccnt 4 retires the older buffer).
//      - S = Q^T K via 4 WMMAs (frags ds_load'ed from LDS)
//      - online softmax (shfl_xor butterflies in the 16-lane C-layout groups)
//      - P transposed C->A layout through LDS; O += P V^T via 4 WMMAs.
// ---------------------------------------------------------------------------
__global__ void __launch_bounds__(128) k_attn(const _Float16* __restrict__ q,
                                              const _Float16* __restrict__ kk,
                                              const _Float16* __restrict__ vv,
                                              _Float16* __restrict__ ao) {
    __shared__ _Float16 Kl[2][32 * 64];   // [buf][key][d]   4KB each
    __shared__ _Float16 Vl[2][64 * 32];   // [buf][d][key]   4KB each
    __shared__ _Float16 Pl[4][16 * 40];   // per-wave P tile, padded stride

    const int tid  = threadIdx.x;
    const int wave = tid >> 5;
    const int t = blockIdx.x * 4 + wave;                // < 4608
    const int bh = t / 144;
    const int i0 = (t - bh * 144) * 16;
    const int b = bh >> 2, head = bh & 3;
    const int l = tid & 31;
    const int n16 = l & 15;
    const bool hi = l >= 16;
    const int kbA = hi ? 16 : 0;
    const int kbB = hi ? 16 : 0;

    const _Float16* qb = q  + (size_t)bh * HW * HD;
    const _Float16* kb = kk + (size_t)bh * HW * HD;
    const _Float16* vb = vv + (size_t)bh * HD * HW;

    // --- async staging: this thread owns 16B chunks tid and tid+128 of each tile
    const int c0 = tid, c1 = tid + 128;
    const char* gK0 = (const char*)kb + (c0 >> 3) * 128 + (c0 & 7) * 16;  // +j0*128
    const char* gK1 = (const char*)kb + (c1 >> 3) * 128 + (c1 & 7) * 16;
    const char* gV0 = (const char*)vb + (size_t)(c0 >> 2) * (HW * 2) + (c0 & 3) * 16;  // +j0*2
    const char* gV1 = (const char*)vb + (size_t)(c1 >> 2) * (HW * 2) + (c1 & 3) * 16;
    // LDS byte offsets: low 32 bits of the generic shared pointer (flat->LDS
    // truncation, ISA 10.2); hardware adds the wave's LDS_BASE.
    const unsigned lK0 = (unsigned)(size_t)(&Kl[0][0]) + c0 * 16;
    const unsigned lK1 = (unsigned)(size_t)(&Kl[0][0]) + c1 * 16;
    const unsigned lV0 = (unsigned)(size_t)(&Vl[0][0]) + c0 * 16;
    const unsigned lV1 = (unsigned)(size_t)(&Vl[0][0]) + c1 * 16;

    // Q tile as two A fragments (d = 0..31 and 32..63), resident all loop long
    const _Float16* aq = qb + (size_t)(i0 + n16) * HD;
    const v16h aQ0 = cat8(*(const v8h*)(aq + kbA),      *(const v8h*)(aq + kbA + 8));
    const v16h aQ1 = cat8(*(const v8h*)(aq + 32 + kbA), *(const v8h*)(aq + 32 + kbA + 8));

    v8f O0 = {}, O1 = {}, O2 = {}, O3 = {};
    float mrow[8], lrow[8];
#pragma unroll
    for (int rr = 0; rr < 8; ++rr) { mrow[rr] = -1e30f; lrow[rr] = 0.f; }

    _Float16* Pw = Pl[wave];

    // prologue: stage block 0 into buffer 0
    async_cp16(lK0, gK0);
    async_cp16(lK1, gK1);
    async_cp16(lV0, gV0);
    async_cp16(lV1, gV1);

    const int NBLK = HW / 32;   // 72
    for (int jb = 0; jb < NBLK; ++jb) {
        const int cur = jb & 1;
        if (jb + 1 < NBLK) {    // prefetch next block into the other buffer
            const unsigned bo = (cur ^ 1) * (unsigned)(32 * 64 * 2);
            const size_t jK = (size_t)(jb + 1) * 32 * 128;   // bytes
            const size_t jV = (size_t)(jb + 1) * 32 * 2;     // bytes
            async_cp16(lK0 + bo, gK0 + jK);
            async_cp16(lK1 + bo, gK1 + jK);
            async_cp16(lV0 + bo, gV0 + jV);
            async_cp16(lV1 + bo, gV1 + jV);
            asm volatile("s_wait_asynccnt 0x4" ::: "memory");  // older 4 done (in-order)
        } else {
            asm volatile("s_wait_asynccnt 0x0" ::: "memory");
        }
        __syncthreads();        // current buffer visible to all 4 waves

        const _Float16* Kc = &Kl[cur][0];
        const _Float16* Vc = &Vl[cur][0];
        const _Float16* kr0 = Kc + (size_t)n16 * 64;          // key sub-tile 0
        const _Float16* kr1 = Kc + (size_t)(16 + n16) * 64;   // key sub-tile 1
        v8f S0 = {}, S1 = {};
        S0 = wmma_f16(aQ0, *(const v16h*)(kr0 + kbB),      S0);
        S0 = wmma_f16(aQ1, *(const v16h*)(kr0 + 32 + kbB), S0);
        S1 = wmma_f16(aQ0, *(const v16h*)(kr1 + kbB),      S1);
        S1 = wmma_f16(aQ1, *(const v16h*)(kr1 + 32 + kbB), S1);

#pragma unroll
        for (int rr = 0; rr < 8; ++rr) {
            float s0 = S0[rr], s1 = S1[rr];
            float mx = fmaxf(s0, s1);
            mx = fmaxf(mx, __shfl_xor(mx, 1, 32));
            mx = fmaxf(mx, __shfl_xor(mx, 2, 32));
            mx = fmaxf(mx, __shfl_xor(mx, 4, 32));
            mx = fmaxf(mx, __shfl_xor(mx, 8, 32));
            const float mn = fmaxf(mrow[rr], mx);
            const float corr = __expf(mrow[rr] - mn);
            const float p0 = __expf(s0 - mn);
            const float p1 = __expf(s1 - mn);
            float rs = p0 + p1;
            rs += __shfl_xor(rs, 1, 32);
            rs += __shfl_xor(rs, 2, 32);
            rs += __shfl_xor(rs, 4, 32);
            rs += __shfl_xor(rs, 8, 32);
            lrow[rr] = lrow[rr] * corr + rs;
            mrow[rr] = mn;
            O0[rr] *= corr; O1[rr] *= corr; O2[rr] *= corr; O3[rr] *= corr;
            const int row = rr + (hi ? 8 : 0);
            Pw[row * 40 + n16]      = (_Float16)p0;
            Pw[row * 40 + 16 + n16] = (_Float16)p1;
        }
        // P write->read is wave-private LDS: DS ops from one wave stay in order.
        const _Float16* pr = Pw + n16 * 40;
        const v16h aP = cat8(*(const v8h*)(pr + kbA), *(const v8h*)(pr + kbA + 8));
        const _Float16* vr = Vc + (size_t)n16 * 32 + kbB;
        O0 = wmma_f16(aP, *(const v16h*)(vr),           O0);
        O1 = wmma_f16(aP, *(const v16h*)(vr + 16 * 32), O1);
        O2 = wmma_f16(aP, *(const v16h*)(vr + 32 * 32), O2);
        O3 = wmma_f16(aP, *(const v16h*)(vr + 48 * 32), O3);
        __syncthreads();        // everyone done reading before buffer reuse
    }

#pragma unroll
    for (int rr = 0; rr < 8; ++rr) {
        const float inv = 1.0f / lrow[rr];
        const int row = i0 + rr + (hi ? 8 : 0);
        const size_t base = ((size_t)(b * HW) + row) * C + head * HD + n16;
        ao[base]      = (_Float16)(O0[rr] * inv);
        ao[base + 16] = (_Float16)(O1[rr] * inv);
        ao[base + 32] = (_Float16)(O2[rr] * inv);
        ao[base + 48] = (_Float16)(O3[rr] * inv);
    }
}

// ---------------------------------------------------------------------------
// 6) Proj GEMM + bias + residual: out = x + W(256x256) x AO(256x2304) + b
// ---------------------------------------------------------------------------
__global__ void __launch_bounds__(256) k_proj(const _Float16* __restrict__ wh,
                                              const float* __restrict__ bias,
                                              const _Float16* __restrict__ ao,
                                              const float* __restrict__ x,
                                              float* __restrict__ out) {
    const int t = blockIdx.x * 8 + (threadIdx.x >> 5);   // < 18432
    const int b = t / (16 * 144);
    const int r = t - b * 16 * 144;
    const int o0 = (r / 144) * 16;
    const int p0 = (r % 144) * 16;
    const int l = threadIdx.x & 31;
    const int n16 = l & 15;
    const bool hi = l >= 16;
    const int kbA = hi ? 16 : 0;
    const int kbB = hi ? 16 : 0;

    const _Float16* xb = ao + ((size_t)(b * HW + p0 + n16) << 8);
    v8f acc = {};
    for (int k0 = 0; k0 < C; k0 += 32) {
        const _Float16* ap = wh + (size_t)(o0 + n16) * C + k0 + kbA;
        v16h a = cat8(*(const v8h*)ap, *(const v8h*)(ap + 8));
        v16h bb = *(const v16h*)(xb + k0 + kbB);
        acc = wmma_f16(a, bb, acc);
    }
    const int rowoff = hi ? 8 : 0;
#pragma unroll
    for (int rr = 0; rr < 8; ++rr) {
        const int o = o0 + rr + rowoff;
        const size_t idx = ((size_t)(b * C) + o) * HW + p0 + n16;
        out[idx] = x[idx] + acc[rr] + bias[o];
    }
}

// ---------------------------------------------------------------------------
extern "C" void kernel_launch(void* const* d_in, const int* in_sizes, int n_in,
                              void* d_out, int out_size, void* d_ws, size_t ws_size,
                              hipStream_t stream) {
    (void)in_sizes; (void)n_in; (void)out_size; (void)ws_size;
    const float* x      = (const float*)d_in[0];
    const float* gn_w   = (const float*)d_in[1];
    const float* gn_b   = (const float*)d_in[2];
    const float* qkv_w  = (const float*)d_in[3];
    const float* qkv_b  = (const float*)d_in[4];
    const float* proj_w = (const float*)d_in[5];
    const float* proj_b = (const float*)d_in[6];
    float* out = (float*)d_out;

    char* ws = (char*)d_ws;
    size_t off = 0;
    auto carve = [&](size_t bytes) -> char* {
        char* p = ws + off;
        off += (bytes + 255) & ~(size_t)255;
        return p;
    };
    float*    mean    = (float*)carve(256 * sizeof(float));
    float*    rstd    = (float*)carve(256 * sizeof(float));
    _Float16* qkvw16  = (_Float16*)carve((size_t)768 * 256 * 2);
    _Float16* projw16 = (_Float16*)carve((size_t)256 * 256 * 2);
    _Float16* xn16    = (_Float16*)carve((size_t)BATCH * HW * C * 2);         // [b][p][c]
    _Float16* q16     = (_Float16*)carve((size_t)BATCH * NH * HW * HD * 2);   // [bh][p][d]
    _Float16* k16     = (_Float16*)carve((size_t)BATCH * NH * HW * HD * 2);   // [bh][p][d]
    _Float16* v16     = (_Float16*)carve((size_t)BATCH * NH * HD * HW * 2);   // [bh][d][p]
    _Float16* ao16    = (_Float16*)carve((size_t)BATCH * HW * C * 2);         // [b][p][c]

    k_gn_stats<<<256, 256, 0, stream>>>(x, mean, rstd);
    k_gn_norm<<<(BATCH * C * HW) / 256, 256, 0, stream>>>(x, mean, rstd, gn_w, gn_b, xn16);
    k_w2h<<<(768 * 256) / 256, 256, 0, stream>>>(qkv_w, qkvw16, 768 * 256);
    k_w2h<<<(256 * 256) / 256, 256, 0, stream>>>(proj_w, projw16, 256 * 256);
    k_qkv<<<6912, 256, 0, stream>>>(qkvw16, qkv_b, xn16, q16, k16, v16);
    k_attn<<<1152, 128, 0, stream>>>(q16, k16, v16, ao16);
    k_proj<<<2304, 256, 0, stream>>>(projw16, proj_b, ao16, x, out);
}
